// GLALayer_62302795596601
// MI455X (gfx1250) — compile-verified, matrix operand order checked by reading, not run on hardware
//
#include <hip/hip_runtime.h>
#include <hip/hip_bf16.h>
#include <math.h>

// ---------------------------------------------------------------------------
// GLA layer for MI455X (gfx1250, wave32):
//   phase 1: 4x projection GEMMs, 128x128 tiles (fp32->bf16 WMMA)  [wmma]
//   phase 2: l2norm(q), l2norm(k), sigmoid(g)                      [valu]
//   phase 3: chunked gated-linear-attention scan, chunk L=32, all
//            chunk matmuls via v_wmma_f32_16x16x32_bf16            [wmma]
//   phase 4: output GEMM o @ Wo -> d_out                           [wmma]
// ---------------------------------------------------------------------------

typedef __bf16 bf16;
typedef __attribute__((ext_vector_type(16))) __bf16 v16bf;
typedef __attribute__((ext_vector_type(8)))  float  v8f;

#define B_    4
#define T_    2048
#define H_    8
#define DK_   128
#define DHID  1024          // H*DK = H*DV
#define MROWS (B_ * T_)     // 8192
#define LCH   32            // chunk length
#define DVC   64            // DV columns per block (DV split in 2)
#define NCH   (T_ / LCH)    // 64 sequential chunks

__device__ __forceinline__ v8f wmma_bf16(v16bf a, v16bf b, v8f c) {
  // (neg_a, A, neg_b, B, c_mod, C, reuse_a, reuse_b)
  return __builtin_amdgcn_wmma_f32_16x16x32_bf16(false, a, false, b,
                                                 (short)0, c, false, false);
}

// Load a 16x32 (rows x K) bf16 fragment from row-major LDS where K is the
// fastest (contiguous) index.  Works for A operands, and for B operands when
// the B tile is staged TRANSPOSED ([n][k]).  Layout per ISA 7.12.2:
// lane = 16*h + m ; elems 0..7 -> K = 8h+e ; elems 8..15 -> K = 16+8h+e.
__device__ __forceinline__ v16bf load_frag(const bf16* base, int ld,
                                           int row0, int k0) {
  const int lane = threadIdx.x & 31;
  const int m = lane & 15;
  const int h = lane >> 4;
  const bf16* p = base + (row0 + m) * ld + k0 + 8 * h;
  v16bf out;
#pragma unroll
  for (int e = 0; e < 8; ++e) out[e] = p[e];
  p += 16;
#pragma unroll
  for (int e = 0; e < 8; ++e) out[8 + e] = p[e];
  return out;
}

// ---------------------------------------------------------------------------
// Tiled GEMM: Y[M,N] = X[M,K] @ W[K,N], fp32 in/out, bf16 WMMA math.
// Block 256 threads = 8 waves, 128x128 tile, K staged 32 at a time.
// Wave w owns m-tile w (rows 16w..16w+15) and all 8 n-tiles -> one shared
// A fragment per K-step feeds 8 WMMAs (high FLOP / LDS-byte ratio).
// Next K panel is prefetched (global_prefetch_b8) while the current one
// is being consumed.
// ---------------------------------------------------------------------------
#define TM 128
#define TN 128
#define TK 32

__global__ __launch_bounds__(256) void gemm_bf16wmma(
    const float* __restrict__ X, const float* __restrict__ W,
    float* __restrict__ Y, int M, int N, int K) {
  __shared__ bf16 sX[TM][TK];    // [m][k]   8 KB
  __shared__ bf16 sWt[TN][TK];   // [n][k]   8 KB (transposed)

  const int tid  = threadIdx.x;
  const int wave = tid >> 5;
  const int lane = tid & 31;
  const int m_blk = blockIdx.y * TM;
  const int n_blk = blockIdx.x * TN;

  v8f acc[8];
#pragma unroll
  for (int j = 0; j < 8; ++j) acc[j] = (v8f){};

  for (int k0 = 0; k0 < K; k0 += TK) {
    __syncthreads();
    for (int idx = tid; idx < TM * TK; idx += 256) {
      int m = idx >> 5, k = idx & 31;
      const float* gp = &X[(size_t)(m_blk + m) * K + k0 + k];
      sX[m][k] = (bf16)(*gp);
      if (k0 + TK < K) __builtin_prefetch(gp + TK, 0, 1);   // next K panel
    }
    for (int idx = tid; idx < TK * TN; idx += 256) {
      int k = idx >> 7, n = idx & 127;
      const float* gp = &W[(size_t)(k0 + k) * N + n_blk + n];
      sWt[n][k] = (bf16)(*gp);
      if (k0 + TK < K) __builtin_prefetch(gp + (size_t)TK * N, 0, 1);
    }
    __syncthreads();

    v16bf a = load_frag(&sX[0][0], TK, wave * 16, 0);
#pragma unroll
    for (int j = 0; j < 8; ++j) {
      v16bf b = load_frag(&sWt[0][0], TK, j * 16, 0);
      acc[j] = wmma_bf16(a, b, acc[j]);
    }
  }

  const int n = lane & 15;
  const int h = lane >> 4;
#pragma unroll
  for (int j = 0; j < 8; ++j) {
#pragma unroll
    for (int r = 0; r < 8; ++r) {
      size_t row = (size_t)(m_blk + wave * 16 + r + 8 * h) * N;
      Y[row + n_blk + j * 16 + n] = acc[j][r];
    }
  }
}

// ---------------------------------------------------------------------------
// L2-normalize rows of length 128 in place (one wave per row).
// ---------------------------------------------------------------------------
__global__ __launch_bounds__(256) void l2norm_rows(float* __restrict__ q,
                                                   int nrows) {
  int row = blockIdx.x * 8 + (threadIdx.x >> 5);
  if (row >= nrows) return;
  int lane = threadIdx.x & 31;
  float* p = q + (size_t)row * DK_;
  float v[4]; float ss = 0.f;
#pragma unroll
  for (int i = 0; i < 4; ++i) { v[i] = p[lane + 32 * i]; ss += v[i] * v[i]; }
#pragma unroll
  for (int off = 16; off; off >>= 1) ss += __shfl_xor(ss, off, 32);
  float inv = 1.f / fmaxf(sqrtf(ss), 1e-12f);
#pragma unroll
  for (int i = 0; i < 4; ++i) p[lane + 32 * i] = v[i] * inv;
}

__global__ __launch_bounds__(256) void sigmoid_inplace(float* __restrict__ g,
                                                       int n) {
  int base = blockIdx.x * 1024 + threadIdx.x;
#pragma unroll
  for (int i = 0; i < 4; ++i) {
    int idx = base + i * 256;
    if (idx < n) g[idx] = 1.f / (1.f + expf(-g[idx]));
  }
}

// ---------------------------------------------------------------------------
// Chunked GLA scan. One block per (b, h, jseg): state S is [DK_ x DVC].
//   per chunk (L=32):
//     lam_t  = cumprod(alpha)          (column scan, 128 threads)
//     q~ = q*lam, k^ = k/lam           (bf16 in LDS)
//     O  = q~ @ S_prev  +  mask(q~ @ k^T) @ V
//     S  = diag(lamL) S + (k^ * lamL)^T @ V
// S lives in WMMA accumulator registers (wave w owns i-tile w, j-tiles 0..3);
// a bf16 transposed copy sSt[j][i] is kept in LDS as the B operand of q~@S.
// ---------------------------------------------------------------------------
__global__ __launch_bounds__(256) void gla_chunk_kernel(
    const float* __restrict__ qn, const float* __restrict__ kn,
    const float* __restrict__ vv, const float* __restrict__ al,
    float* __restrict__ ob) {
  __shared__ bf16 sQ[LCH][DK_];    // q~ [t][i]
  __shared__ bf16 sK[LCH][DK_];    // k^ [t][i]
  __shared__ bf16 sVt[DVC][LCH];   // v  transposed [j][t]
  __shared__ bf16 sSt[DVC][DK_];   // S  transposed [j][i] (bf16 copy)
  __shared__ bf16 sA[LCH][LCH];    // masked attention [t][s]
  __shared__ float sLamL[DK_];

  const int tid  = threadIdx.x;
  const int wave = tid >> 5;
  const int lane = tid & 31;
  const int n_   = lane & 15;
  const int hh   = lane >> 4;

  const int bh   = blockIdx.x >> 1;
  const int jseg = blockIdx.x & 1;
  const int b    = bh >> 3;
  const int h    = bh & 7;

  const size_t rowbase = (size_t)b * T_;
  const int colqk = h * DK_;
  const int colv  = h * DK_ + jseg * DVC;

  v8f Sacc[4];
#pragma unroll
  for (int j = 0; j < 4; ++j) Sacc[j] = (v8f){};
  for (int idx = tid; idx < DVC * DK_; idx += 256) (&sSt[0][0])[idx] = (bf16)0.f;
  __syncthreads();

  const int mt = (wave & 1) * 16;   // O row tile
  const int jt = wave >> 1;         // O col tile (0..3)

  for (int c = 0; c < NCH; ++c) {
    const int t0 = c * LCH;

    // ---- cumulative-decay scan + staging (fp32 math, bf16 store) ----
    if (tid < DK_) {
      const int i = tid;
      float lam = 1.f;
#pragma unroll 4
      for (int t = 0; t < LCH; ++t) {
        size_t g = (rowbase + t0 + t) * (size_t)DHID + colqk + i;
        lam *= al[g];
        sQ[t][i] = (bf16)(qn[g] * lam);
        sK[t][i] = (bf16)(kn[g] / lam);
      }
      sLamL[i] = lam;
    }
    for (int idx = tid; idx < DVC * LCH; idx += 256) {
      int j = idx >> 5, t = idx & (LCH - 1);
      sVt[j][t] = (bf16)vv[(rowbase + t0 + t) * (size_t)DHID + colv + j];
    }
    __syncthreads();

    // ---- inter-chunk: O += q~ @ S_prev  (K = DK_) ----
    v8f Oacc = (v8f){};
#pragma unroll
    for (int k0 = 0; k0 < DK_; k0 += 32) {
      v16bf a   = load_frag(&sQ[0][0],  DK_, mt,      k0);
      v16bf bS  = load_frag(&sSt[0][0], DK_, jt * 16, k0);
      Oacc = wmma_bf16(a, bS, Oacc);
    }

    // ---- intra-chunk attention A = mask(q~ @ k^T), waves 0..3 ----
    if (wave < 4) {
      const int amt = (wave & 1) * 16;
      const int ant = (wave >> 1) * 16;
      v8f Acc = (v8f){};
#pragma unroll
      for (int k0 = 0; k0 < DK_; k0 += 32) {
        v16bf a  = load_frag(&sQ[0][0], DK_, amt, k0);
        v16bf bk = load_frag(&sK[0][0], DK_, ant, k0);
        Acc = wmma_bf16(a, bk, Acc);
      }
#pragma unroll
      for (int r = 0; r < 8; ++r) {
        int tt = amt + r + 8 * hh;
        int ss = ant + n_;
        sA[tt][ss] = (tt >= ss) ? (bf16)Acc[r] : (bf16)0.f;
      }
    }
    __syncthreads();

    // ---- intra-chunk: O += A @ V  (K = LCH) ----
    {
      v16bf a  = load_frag(&sA[0][0],  LCH, mt,      0);
      v16bf bv = load_frag(&sVt[0][0], LCH, jt * 16, 0);
      Oacc = wmma_bf16(a, bv, Oacc);
    }
    // store O tile (fp32)
    {
      float* obase = ob + (rowbase + t0) * (size_t)DHID + colv;
#pragma unroll
      for (int r = 0; r < 8; ++r)
        obase[(size_t)(mt + r + 8 * hh) * DHID + jt * 16 + n_] = Oacc[r];
    }

    // ---- state update: S = diag(lamL) S + (k^ * lamL)^T @ V ----
    float lamrow[8];
#pragma unroll
    for (int r = 0; r < 8; ++r) lamrow[r] = sLamL[16 * wave + r + 8 * hh];

    const int iglob = 16 * wave + n_;          // A-operand row = k-index i
    const float lsc = sLamL[iglob];
    v16bf af;
#pragma unroll
    for (int e = 0; e < 8; ++e) {
      af[e]     = (bf16)((float)sK[8 * hh + e][iglob] * lsc);
      af[8 + e] = (bf16)((float)sK[16 + 8 * hh + e][iglob] * lsc);
    }
#pragma unroll
    for (int j = 0; j < 4; ++j) {
#pragma unroll
      for (int r = 0; r < 8; ++r) Sacc[j][r] *= lamrow[r];
      v16bf bv = load_frag(&sVt[0][0], LCH, j * 16, 0);
      Sacc[j] = wmma_bf16(af, bv, Sacc[j]);
    }
    __syncthreads();                 // O_inter readers done with old sSt

    // publish bf16 S^T for next chunk's q~@S
    const int i0 = 16 * wave + 8 * hh;
#pragma unroll
    for (int j = 0; j < 4; ++j)
#pragma unroll
      for (int r = 0; r < 8; ++r)
        sSt[j * 16 + n_][i0 + r] = (bf16)Sacc[j][r];
    __syncthreads();
  }
}

// ---------------------------------------------------------------------------
extern "C" void kernel_launch(void* const* d_in, const int* in_sizes, int n_in,
                              void* d_out, int out_size, void* d_ws, size_t ws_size,
                              hipStream_t stream) {
  (void)in_sizes; (void)n_in; (void)out_size; (void)ws_size;
  const float* x  = (const float*)d_in[0];
  const float* Wq = (const float*)d_in[1];
  const float* Wk = (const float*)d_in[2];
  const float* Wv = (const float*)d_in[3];
  const float* Wg = (const float*)d_in[4];
  const float* Wo = (const float*)d_in[5];
  float* out = (float*)d_out;

  const size_t NPROJ = (size_t)MROWS * DHID;   // 8.39M floats each
  float* qn = (float*)d_ws;
  float* kn = qn + NPROJ;
  float* vv = kn + NPROJ;
  float* al = vv + NPROJ;
  float* ob = al + NPROJ;

  dim3 blk(256);
  dim3 gg(DHID / TN, MROWS / TM);   // 8 x 64

  // phase 1: projections
  gemm_bf16wmma<<<gg, blk, 0, stream>>>(x, Wq, qn, MROWS, DHID, DHID);
  gemm_bf16wmma<<<gg, blk, 0, stream>>>(x, Wk, kn, MROWS, DHID, DHID);
  gemm_bf16wmma<<<gg, blk, 0, stream>>>(x, Wv, vv, MROWS, DHID, DHID);
  gemm_bf16wmma<<<gg, blk, 0, stream>>>(x, Wg, al, MROWS, DHID, DHID);

  // phase 2: normalization / gate
  int nrows = MROWS * H_;                       // 65536 rows of 128
  l2norm_rows<<<nrows / 8, blk, 0, stream>>>(qn, nrows);
  l2norm_rows<<<nrows / 8, blk, 0, stream>>>(kn, nrows);
  sigmoid_inplace<<<(int)(NPROJ / 1024), blk, 0, stream>>>(al, (int)NPROJ);

  // phase 3: chunked GLA scan  (B*H*2 = 64 blocks)
  gla_chunk_kernel<<<B_ * H_ * 2, blk, 0, stream>>>(qn, kn, vv, al, ob);

  // phase 4: output projection
  gemm_bf16wmma<<<gg, blk, 0, stream>>>(ob, Wo, out, MROWS, DHID, DHID);
}
